// SFCNModel_35785667510311
// MI455X (gfx1250) — matrix operand order repaired and checked by reading, not compiled, not run on hardware
//
#include <hip/hip_runtime.h>

typedef __attribute__((ext_vector_type(16))) _Float16 v16h;
typedef __attribute__((ext_vector_type(8)))  _Float16 v8h;
typedef __attribute__((ext_vector_type(8)))  float    v8f;

#define STEPS   100
#define THRESH  1.0f
#define DECAY   0.99f

#define BDIM    2048
#define D_IN    784
#define D_PAD   800      // 25 * 32
#define H1      2048
#define H2      1024
#define OUTN    10
#define OUT_PAD 128      // padded N for layer 3

#define BM 128
#define BN 128
#define BK 32

// ------------------------------------------------------ CDNA5 async helpers
__device__ __forceinline__ void async_load_b128_to_lds(unsigned lds_off,
                                                       const void* gptr) {
    unsigned long long ga = (unsigned long long)gptr;
    // GLOBAL_LOAD_ASYNC_TO_LDS_B128: per-lane 16B, memory -> LDS, ASYNCcnt.
    asm volatile("global_load_async_to_lds_b128 %0, %1, off"
                 :
                 : "v"(lds_off), "v"(ga)
                 : "memory");
}

__device__ __forceinline__ void wait_async0() {
#if __has_builtin(__builtin_amdgcn_s_wait_asynccnt)
    __builtin_amdgcn_s_wait_asynccnt(0);
#else
    asm volatile("s_wait_asynccnt 0x0" ::: "memory");
#endif
}

// ---------------------------------------------------------------- PRNG
__device__ __forceinline__ float prng01(unsigned step, unsigned idx) {
    unsigned long long z = ((unsigned long long)(step + 1) << 32) ^ (unsigned long long)(idx * 2654435761u);
    z = (z ^ (z >> 30)) * 0xBF58476D1CE4E5B9ull;
    z = (z ^ (z >> 27)) * 0x94D049BB133111EBull;
    z = z ^ (z >> 31);
    return (float)((unsigned)(z >> 40)) * (1.0f / 16777216.0f);
}

// ------------------------------------------------- input spike encoding
__global__ __launch_bounds__(256) void spike_gen(const float* __restrict__ x,
                                                 _Float16* __restrict__ s0,
                                                 int step) {
    int i = blockIdx.x * blockDim.x + threadIdx.x;
    if (i >= BDIM * D_PAD) return;
    int b = i / D_PAD, k = i % D_PAD;
    _Float16 v = (_Float16)0.0f;
    if (k < D_IN) {
        float xv   = x[b * D_IN + k];
        float sgn  = (xv > 0.0f) ? 1.0f : ((xv < 0.0f) ? -1.0f : 0.0f);
        float prob = fabsf(xv) * 0.5f;
        float r    = prng01((unsigned)step, (unsigned)(b * D_IN + k));
        v = (prob > r) ? (_Float16)sgn : (_Float16)0.0f;
    }
    s0[i] = v;
}

// ----------------------------------- weight convert + transpose (f32->f16)
// Wt[n*Kp + k] = W[k*N + n], zero padded
__global__ __launch_bounds__(256) void prep_wt(const float* __restrict__ W,
                                               _Float16* __restrict__ Wt,
                                               int K, int N, int Kp, int Np) {
    int i = blockIdx.x * blockDim.x + threadIdx.x;
    if (i >= Np * Kp) return;
    int n = i / Kp, k = i % Kp;
    float v = (k < K && n < N) ? W[k * N + n] : 0.0f;
    Wt[i] = (_Float16)v;
}

__global__ void pad_bias(const float* __restrict__ b, float* __restrict__ bp,
                         int n, int np) {
    int i = blockIdx.x * blockDim.x + threadIdx.x;
    if (i < np) bp[i] = (i < n) ? b[i] : 0.0f;
}

// ------------------------------------------------- fused GEMM + LIF + LF
// C[M x Np] = A[M x Kp] * Bt[Np x Kp]^T ; then LIF + low-pass + spike count.
// 128x128 block tile, BK=32, 256 threads = 8 waves, wave -> 32x64 sub-tile.
// Double-buffered LDS filled by GLOBAL_LOAD_ASYNC_TO_LDS_B128 (ASYNCcnt),
// overlapping the next K-tile's copy with the current tile's WMMAs.
__global__ __launch_bounds__(256) void gemm_lif(
    const _Float16* __restrict__ A,    // [M x Kp]  spikes (row major)
    const _Float16* __restrict__ Bt,   // [Np x Kp] weights, K-major
    const float*    __restrict__ bias, // [Np]
    float* __restrict__ mem,           // [M x Np]  membrane state (in/out)
    float* __restrict__ lf,            // [M x Np]  low-pass state (in/out)
    float* __restrict__ tot,           // [M x Np]  spike counts   (in/out)
    _Float16* __restrict__ Sout,       // [M x Np]  output spikes (next layer A)
    int Kp, int Np) {

    __shared__ __align__(16) _Float16 As[2][BM * BK];
    __shared__ __align__(16) _Float16 Bs[2][BN * BK];

    const int tid  = threadIdx.x;
    const int lane = tid & 31;
    const int wv   = tid >> 5;
    const int half = lane >> 4;
    const int l15  = lane & 15;

    const int rowBase = blockIdx.y * BM;
    const int colBase = blockIdx.x * BN;
    const int waveRow = (wv & 3) * 32;
    const int waveCol = (wv >> 2) * 64;

    v8f acc[2][4];
#pragma unroll
    for (int i = 0; i < 2; ++i)
#pragma unroll
        for (int j = 0; j < 4; ++j)
#pragma unroll
            for (int r = 0; r < 8; ++r) acc[i][j][r] = 0.0f;

    // Per-thread staging chunks: tile is 128 rows x 4 chunks of 16B.
    const int cr = tid >> 2;     // rows handled: cr and cr+64
    const int cc = (tid & 3) * 8;  // f16 column offset of 16B chunk

    auto issue_tile = [&](int kt, int buf) {
#pragma unroll
        for (int i = 0; i < 2; ++i) {
            int r = cr + i * 64;
            async_load_b128_to_lds(
                (unsigned)(size_t)&As[buf][r * BK + cc],
                A + (size_t)(rowBase + r) * Kp + kt * BK + cc);
            async_load_b128_to_lds(
                (unsigned)(size_t)&Bs[buf][r * BK + cc],
                Bt + (size_t)(colBase + r) * Kp + kt * BK + cc);
        }
    };

    const int nk = Kp / BK;
    issue_tile(0, 0);

    for (int kt = 0; kt < nk; ++kt) {
        const int cur = kt & 1;
        wait_async0();      // this wave's contributions to buffer `cur` landed
        __syncthreads();    // all waves' contributions visible

        if (kt + 1 < nk) issue_tile(kt + 1, cur ^ 1);  // overlap with WMMAs

        // A fragment (16x32 f16): lane = M%16, half selects K-interleave group.
        v16h af[2], bf[4];
#pragma unroll
        for (int i = 0; i < 2; ++i) {
            const v8h* p = (const v8h*)&As[cur][(waveRow + i * 16 + l15) * BK];
            v8h lo = p[half];        // K =      8*half .. 8*half+7
            v8h hi = p[half + 2];    // K = 16 + 8*half .. +7
            af[i] = __builtin_shufflevector(lo, hi, 0,1,2,3,4,5,6,7,8,9,10,11,12,13,14,15);
        }
        // B fragment (32x16 f16): lane = N%16, half selects K-interleave group.
#pragma unroll
        for (int j = 0; j < 4; ++j) {
            const v8h* p = (const v8h*)&Bs[cur][(waveCol + j * 16 + l15) * BK];
            v8h lo = p[half];
            v8h hi = p[half + 2];
            bf[j] = __builtin_shufflevector(lo, hi, 0,1,2,3,4,5,6,7,8,9,10,11,12,13,14,15);
        }

#pragma unroll
        for (int j = 0; j < 4; ++j) {
#pragma unroll
            for (int i = 0; i < 2; ++i) {
                acc[i][j] = __builtin_amdgcn_wmma_f32_16x16x32_f16(
                    /*neg_a=*/false, af[i], /*neg_b=*/false, bf[j],
                    /*c_mod=*/(short)0, acc[i][j],
                    /*reuse_a=*/false, /*reuse_b=*/false);
            }
        }
        __syncthreads();    // reads of buffer `cur` done before it is refilled
    }

    // ---- fused LIF / LF / count epilogue
    // C-fragment layout: VGPR r, lanes 0-15 -> M=r, lanes 16-31 -> M=r+8; N=lane&15.
#pragma unroll
    for (int i = 0; i < 2; ++i) {
        int tr = rowBase + waveRow + i * 16;
#pragma unroll
        for (int j = 0; j < 4; ++j) {
            int tc = colBase + waveCol + j * 16 + l15;
            float bv = bias[tc];
#pragma unroll
            for (int r = 0; r < 8; ++r) {
                int row = tr + r + 8 * half;
                size_t idx = (size_t)row * Np + tc;
                float m  = mem[idx] + acc[i][j][r] + bv;
                bool  fired = (m > THRESH);
                float sp = fired ? 1.0f : 0.0f;
                mem[idx]  = fired ? 0.0f : m;
                lf[idx]   = DECAY * lf[idx] + sp;
                tot[idx] += sp;
                Sout[idx] = (_Float16)sp;
            }
        }
    }
}

// ------------------------------- gather layer-3 valid columns into d_out
__global__ void finalize(const float* __restrict__ mem3p,
                         const float* __restrict__ tot3p,
                         const float* __restrict__ lf3p,
                         float* __restrict__ out_mem3,
                         float* __restrict__ out_tot3,
                         float* __restrict__ out_lf3) {
    int i = blockIdx.x * blockDim.x + threadIdx.x;
    if (i >= BDIM * OUTN) return;
    int b = i / OUTN, c = i % OUTN;
    out_mem3[i] = mem3p[b * OUT_PAD + c];
    out_tot3[i] = tot3p[b * OUT_PAD + c];
    out_lf3[i]  = lf3p[b * OUT_PAD + c];
}

// ---------------------------------------------------------------- launch
extern "C" void kernel_launch(void* const* d_in, const int* in_sizes, int n_in,
                              void* d_out, int out_size, void* d_ws, size_t ws_size,
                              hipStream_t stream) {
    const float* x  = (const float*)d_in[0];
    const float* W1 = (const float*)d_in[1];
    const float* b1 = (const float*)d_in[2];
    const float* W2 = (const float*)d_in[3];
    const float* b2 = (const float*)d_in[4];
    const float* W3 = (const float*)d_in[5];
    const float* b3 = (const float*)d_in[6];
    float* out = (float*)d_out;

    // workspace carve-out
    char*  ws  = (char*)d_ws;
    size_t off = 0;
    auto carve = [&](size_t bytes) -> char* {
        char* p = ws + off;
        off += (bytes + 255) & ~(size_t)255;
        return p;
    };
    _Float16* W1t   = (_Float16*)carve((size_t)H1 * D_PAD * 2);
    _Float16* W2t   = (_Float16*)carve((size_t)H2 * H1 * 2);
    _Float16* W3t   = (_Float16*)carve((size_t)OUT_PAD * H2 * 2);
    float*    b3p   = (float*)   carve(OUT_PAD * 4);
    _Float16* s0    = (_Float16*)carve((size_t)BDIM * D_PAD * 2);
    _Float16* s1    = (_Float16*)carve((size_t)BDIM * H1 * 2);
    _Float16* s2    = (_Float16*)carve((size_t)BDIM * H2 * 2);
    _Float16* s3    = (_Float16*)carve((size_t)BDIM * OUT_PAD * 2);
    float*    mem1  = (float*)   carve((size_t)BDIM * H1 * 4);
    float*    mem2  = (float*)   carve((size_t)BDIM * H2 * 4);
    float*    mem3p = (float*)   carve((size_t)BDIM * OUT_PAD * 4);
    float*    lf3p  = (float*)   carve((size_t)BDIM * OUT_PAD * 4);
    float*    tot3p = (float*)   carve((size_t)BDIM * OUT_PAD * 4);

    // d_out layout: mem3, tot1, tot2, tot3, lf1, lf2, lf3
    size_t o = 0;
    float* o_mem3 = out + o; o += (size_t)BDIM * OUTN;
    float* o_tot1 = out + o; o += (size_t)BDIM * H1;
    float* o_tot2 = out + o; o += (size_t)BDIM * H2;
    float* o_tot3 = out + o; o += (size_t)BDIM * OUTN;
    float* o_lf1  = out + o; o += (size_t)BDIM * H1;
    float* o_lf2  = out + o; o += (size_t)BDIM * H2;
    float* o_lf3  = out + o; o += (size_t)BDIM * OUTN;

    // one-time-per-call prep (deterministic; re-run every call)
    prep_wt<<<((size_t)H1 * D_PAD + 255) / 256, 256, 0, stream>>>(W1, W1t, D_IN, H1, D_PAD, H1);
    prep_wt<<<((size_t)H2 * H1 + 255) / 256, 256, 0, stream>>>(W2, W2t, H1, H2, H1, H2);
    prep_wt<<<((size_t)OUT_PAD * H2 + 255) / 256, 256, 0, stream>>>(W3, W3t, H2, OUTN, H2, OUT_PAD);
    pad_bias<<<1, OUT_PAD, 0, stream>>>(b3, b3p, OUTN, OUT_PAD);

    // zero all state (incl. lf/tot regions living directly in d_out)
    hipMemsetAsync(mem1,  0, (size_t)BDIM * H1 * 4, stream);
    hipMemsetAsync(mem2,  0, (size_t)BDIM * H2 * 4, stream);
    hipMemsetAsync(mem3p, 0, (size_t)BDIM * OUT_PAD * 4, stream);
    hipMemsetAsync(lf3p,  0, (size_t)BDIM * OUT_PAD * 4, stream);
    hipMemsetAsync(tot3p, 0, (size_t)BDIM * OUT_PAD * 4, stream);
    hipMemsetAsync(o_tot1, 0, (size_t)BDIM * H1 * 4, stream);
    hipMemsetAsync(o_tot2, 0, (size_t)BDIM * H2 * 4, stream);
    hipMemsetAsync(o_lf1,  0, (size_t)BDIM * H1 * 4, stream);
    hipMemsetAsync(o_lf2,  0, (size_t)BDIM * H2 * 4, stream);

    dim3 blk(256);
    dim3 g1(H1 / BN,      BDIM / BM);   // 16 x 16
    dim3 g2(H2 / BN,      BDIM / BM);   //  8 x 16
    dim3 g3(OUT_PAD / BN, BDIM / BM);   //  1 x 16

    for (int t = 0; t < STEPS; ++t) {
        spike_gen<<<(BDIM * D_PAD) / 256, blk, 0, stream>>>(x, s0, t);
        gemm_lif<<<g1, blk, 0, stream>>>(s0, W1t, b1,  mem1,  o_lf1, o_tot1, s1, D_PAD, H1);
        gemm_lif<<<g2, blk, 0, stream>>>(s1, W2t, b2,  mem2,  o_lf2, o_tot2, s2, H1,    H2);
        gemm_lif<<<g3, blk, 0, stream>>>(s2, W3t, b3p, mem3p, lf3p,  tot3p,  s3, H2,    OUT_PAD);
    }

    finalize<<<(BDIM * OUTN + 255) / 256, blk, 0, stream>>>(mem3p, tot3p, lf3p,
                                                            o_mem3, o_tot3, o_lf3);
}
